// SurvivalLoss_39118562132536
// MI455X (gfx1250) — matrix operand order repaired and checked by reading, not compiled, not run on hardware
//
#include <hip/hip_runtime.h>
#include <hip/hip_bf16.h>
#include <math.h>

// CDNA5 / gfx1250, wave32. Cox partial-likelihood loss via WMMA:
//   S_i = sum_j exp(p_j - pmax) * [t_j >= t_i]   (B x B masked matvec -> WMMA)
//   loss = -(1/n_events) * sum_i ind_i * (p_i - pmax - log S_i)
//
// Kernel 2 inner loop is software-pipelined: LDS loads for chunk c+1 are
// issued before chunk c is consumed, so the compiler can wait on dscnt<=N
// instead of dscnt==0 and hide LDS latency behind mask VALU + WMMA (which
// co-execute on CDNA5: 16-bit WMMA is tracked as a TRANS op).

typedef __attribute__((ext_vector_type(16))) _Float16 v16h;
typedef __attribute__((ext_vector_type(8)))  float    v8f;

#define B_N        8192
#define TILE_ROWS  16
#define CHUNK_K    32
#define NCHUNK     (B_N / CHUNK_K)     // 256
#define NTILES     (B_N / TILE_ROWS)   // 512
#define MAIN_BLOCKS   64
#define MAIN_THREADS  256              // 8 wave32 waves per block
#define WAVES_PER_BLK (MAIN_THREADS / 32)

// ---------------------------------------------------------------------------
// Kernel 1: pmax = max(pred). Single block, deterministic tree reduction.
// ---------------------------------------------------------------------------
__global__ void __launch_bounds__(1024)
cox_max_kernel(const float* __restrict__ pred, float* __restrict__ wsf) {
    __shared__ float sm[1024];
    const int t = threadIdx.x;
    float m = -INFINITY;
    for (int j = t; j < B_N; j += 1024) m = fmaxf(m, pred[j]);
    sm[t] = m;
    __syncthreads();
    for (int s = 512; s > 0; s >>= 1) {
        if (t < s) sm[t] = fmaxf(sm[t], sm[t + s]);
        __syncthreads();
    }
    if (t == 0) wsf[0] = sm[0];
}

// ---------------------------------------------------------------------------
// Per-chunk register payload for the pipelined WMMA loop.
// ---------------------------------------------------------------------------
struct ChunkData {
    float4 t0, t1, t2, t3;   // 16 f32 times for this lane's A K-indices
    v16h   b;                // 16 f16 exp-values for this lane's B K-indices
};

__device__ __forceinline__ ChunkData
load_chunk(const float* Tlds, const _Float16* Elds, int base, bool lo) {
    ChunkData cd;
    // A k order used here (contiguous, not AMD's table order): lanes 0-15 hold K 0..15,
    // lanes 16-31 hold K 16..31 (matches B).
    const int k0 = base + (lo ? 0  : 16);
    const int k1 = base + (lo ? 8  : 24);
    // B uses the same contiguous order: lanes 0-15 hold K 0..15, lanes 16-31 K 16..31.
    const int kb = base + (lo ? 0 : 16);
    cd.t0 = *(const float4*)&Tlds[k0];
    cd.t1 = *(const float4*)&Tlds[k0 + 4];
    cd.t2 = *(const float4*)&Tlds[k1];
    cd.t3 = *(const float4*)&Tlds[k1 + 4];
    cd.b  = *(const v16h*)&Elds[kb];          // 32B-aligned contiguous run
    return cd;
}

__device__ __forceinline__ v16h build_mask(const ChunkData& cd, float ti) {
    const _Float16 one  = (_Float16)1.0f;
    const _Float16 zero = (_Float16)0.0f;
    v16h a;
    a[0]  = (cd.t0.x >= ti) ? one : zero;
    a[1]  = (cd.t0.y >= ti) ? one : zero;
    a[2]  = (cd.t0.z >= ti) ? one : zero;
    a[3]  = (cd.t0.w >= ti) ? one : zero;
    a[4]  = (cd.t1.x >= ti) ? one : zero;
    a[5]  = (cd.t1.y >= ti) ? one : zero;
    a[6]  = (cd.t1.z >= ti) ? one : zero;
    a[7]  = (cd.t1.w >= ti) ? one : zero;
    a[8]  = (cd.t2.x >= ti) ? one : zero;
    a[9]  = (cd.t2.y >= ti) ? one : zero;
    a[10] = (cd.t2.z >= ti) ? one : zero;
    a[11] = (cd.t2.w >= ti) ? one : zero;
    a[12] = (cd.t3.x >= ti) ? one : zero;
    a[13] = (cd.t3.y >= ti) ? one : zero;
    a[14] = (cd.t3.z >= ti) ? one : zero;
    a[15] = (cd.t3.w >= ti) ? one : zero;
    return a;
}

// ---------------------------------------------------------------------------
// Kernel 2: per-row masked log-sum-exp via v_wmma_f32_16x16x32_f16.
// One wave per 16-row tile; K marches over all 8192 samples in chunks of 32.
// A = risk-set mask (f16 0/1), B = exp(p - pmax) replicated across N columns.
// ---------------------------------------------------------------------------
__global__ void __launch_bounds__(MAIN_THREADS)
cox_logsum_kernel(const float* __restrict__ pred,
                  const unsigned char* __restrict__ ind,
                  const float* __restrict__ gt_time,
                  const float* __restrict__ wsf,      // [0] = pmax
                  float* __restrict__ terms) {        // [B_N] out
    __shared__ __align__(32) float    Tlds[B_N];   // 32 KB
    __shared__ __align__(32) _Float16 Elds[B_N];   // 16 KB

    const float pmax = wsf[0];

    // Stage times (f32) and exp(p - pmax) (f16, in (0,1]) into LDS.
    for (int j = threadIdx.x; j < B_N; j += MAIN_THREADS) {
        Tlds[j] = gt_time[j];
        Elds[j] = (_Float16)expf(pred[j] - pmax);
    }
    __syncthreads();

    const int  lane = threadIdx.x & 31;
    const int  wave = threadIdx.x >> 5;
    const int  tile = blockIdx.x * WAVES_PER_BLK + wave;   // 0..511
    const bool lo   = lane < 16;
    const int  mrow = lane & 15;                            // A-matrix row M
    const float ti  = Tlds[tile * TILE_ROWS + mrow];

    v8f acc = {};
    // Software-pipelined main loop: prefetch chunk c+1 while consuming c.
    ChunkData cur = load_chunk(Tlds, Elds, 0, lo);
    for (int c = 0; c < NCHUNK - 1; ++c) {
        ChunkData nxt = load_chunk(Tlds, Elds, (c + 1) * CHUNK_K, lo);
        const v16h a = build_mask(cur, ti);
        // D = A x B + C : (neg_a, A, neg_b, B, c_mod, C, reuse_a, reuse_b)
        acc = __builtin_amdgcn_wmma_f32_16x16x32_f16(
                  false, a, false, cur.b, (short)0, acc, false, false);
        cur = nxt;
    }
    {   // epilogue: last chunk
        const v16h a = build_mask(cur, ti);
        acc = __builtin_amdgcn_wmma_f32_16x16x32_f16(
                  false, a, false, cur.b, (short)0, acc, false, false);
    }

    // D layout: VGPR r holds row r (lanes 0-15) / row r+8 (lanes 16-31);
    // every column is identical (B replicated), so take column-0 lanes.
    if ((lane & 15) == 0) {
#pragma unroll
        for (int r = 0; r < 8; ++r) {
            const int row = tile * TILE_ROWS + (lo ? r : (r + 8));
            const float S      = acc[r];                 // sum over risk set
            const float logsum = pmax + logf(S);         // finite: diag in set
            terms[row] = ind[row] ? (pred[row] - logsum) : 0.0f;
        }
    }
}

// ---------------------------------------------------------------------------
// Kernel 3: loss = -(sum terms) / (sum indicators). Deterministic tree.
// ---------------------------------------------------------------------------
__global__ void __launch_bounds__(1024)
cox_finalize_kernel(const unsigned char* __restrict__ ind,
                    const float* __restrict__ terms,
                    float* __restrict__ out) {
    __shared__ float s_t[1024];
    __shared__ float s_n[1024];
    const int t = threadIdx.x;
    float st = 0.0f, sn = 0.0f;
    for (int j = t; j < B_N; j += 1024) {
        st += terms[j];
        sn += ind[j] ? 1.0f : 0.0f;
    }
    s_t[t] = st; s_n[t] = sn;
    __syncthreads();
    for (int s = 512; s > 0; s >>= 1) {
        if (t < s) { s_t[t] += s_t[t + s]; s_n[t] += s_n[t + s]; }
        __syncthreads();
    }
    if (t == 0) out[0] = -(s_t[0] / s_n[0]);
}

// ---------------------------------------------------------------------------
extern "C" void kernel_launch(void* const* d_in, const int* in_sizes, int n_in,
                              void* d_out, int out_size, void* d_ws, size_t ws_size,
                              hipStream_t stream) {
    (void)in_sizes; (void)n_in; (void)out_size; (void)ws_size;
    const float*         pred    = (const float*)d_in[0];         // [8192,1] f32
    const unsigned char* gt_ind  = (const unsigned char*)d_in[1]; // [8192] bool
    const float*         gt_time = (const float*)d_in[2];         // [8192] f32
    float* wsf   = (float*)d_ws;       // [0] = pmax
    float* terms = wsf + 256;          // [8192] per-row terms
    float* out   = (float*)d_out;      // scalar f32 loss

    cox_max_kernel<<<dim3(1), dim3(1024), 0, stream>>>(pred, wsf);
    cox_logsum_kernel<<<dim3(MAIN_BLOCKS), dim3(MAIN_THREADS), 0, stream>>>(
        pred, gt_ind, gt_time, wsf, terms);
    cox_finalize_kernel<<<dim3(1), dim3(1024), 0, stream>>>(gt_ind, terms, out);
}
